// StemGNN_50921132261797
// MI455X (gfx1250) — matrix operand order repaired and checked by reading, not compiled, not run on hardware
//
#include <hip/hip_runtime.h>
#include <hip/hip_bf16.h>

typedef __attribute__((ext_vector_type(16))) _Float16 v16h;
typedef __attribute__((ext_vector_type(8)))  float    v8f;

#define BSZ 64
#define SSZ 128
#define HSZ 64
#define NSZ (BSZ * SSZ)        // 8192 nodes
#define EPSV 1e-5f
#define CHUNKS 4
#define CHLEN (NSZ / CHUNKS)   // 2048 j per split-K chunk

union AFrag { v16h v; _Float16 e[16]; };
union DFrag { v8f  v; float     e[8];  };

// ---------------------------------------------------------------------------
// Kernel 1: conv1d(1->64,k3)+BN+ReLU, conv1d(64->64,k3)+BN+ReLU  -> xf[N][64]
// ---------------------------------------------------------------------------
__global__ __launch_bounds__(256)
void k_feat(const float* __restrict__ x,
            const float* __restrict__ c1w, const float* __restrict__ c1b,
            const float* __restrict__ g1,  const float* __restrict__ b1,
            const float* __restrict__ m1,  const float* __restrict__ v1,
            const float* __restrict__ c2w, const float* __restrict__ c2b,
            const float* __restrict__ g2,  const float* __restrict__ b2,
            const float* __restrict__ m2,  const float* __restrict__ v2,
            float* __restrict__ xf) {
    __shared__ float xs[SSZ + 2];
    __shared__ float h1[HSZ][SSZ + 2];
    __shared__ float sc1[HSZ], sh1[HSZ], sc2[HSZ], sh2[HSZ];
    const int tid = threadIdx.x;
    const int b   = blockIdx.x;

    if (tid < SSZ) xs[tid + 1] = x[b * SSZ + tid];
    if (tid == 0) { xs[0] = 0.f; xs[SSZ + 1] = 0.f; }
    if (tid < HSZ) {
        float inv = g1[tid] * rsqrtf(v1[tid] + EPSV);
        sc1[tid] = inv;
        sh1[tid] = c1b[tid] * inv + b1[tid] - m1[tid] * inv;
        float inv2 = g2[tid] * rsqrtf(v2[tid] + EPSV);
        sc2[tid] = inv2;
        sh2[tid] = c2b[tid] * inv2 + b2[tid] - m2[tid] * inv2;
        h1[tid][0] = 0.f; h1[tid][SSZ + 1] = 0.f;
    }
    __syncthreads();

    for (int e = tid; e < HSZ * SSZ; e += 256) {
        int c = e >> 7, s = e & (SSZ - 1);
        float y = c1w[c * 3 + 0] * xs[s] + c1w[c * 3 + 1] * xs[s + 1] +
                  c1w[c * 3 + 2] * xs[s + 2];
        y = y * sc1[c] + sh1[c];
        h1[c][s + 1] = fmaxf(y, 0.f);
    }
    __syncthreads();

    for (int e = tid; e < HSZ * SSZ; e += 256) {
        int c = e >> 7, s = e & (SSZ - 1);
        float acc = 0.f;
        const float* wr = c2w + c * HSZ * 3;
        #pragma unroll 8
        for (int ci = 0; ci < HSZ; ++ci) {
            acc += wr[ci * 3 + 0] * h1[ci][s] +
                   wr[ci * 3 + 1] * h1[ci][s + 1] +
                   wr[ci * 3 + 2] * h1[ci][s + 2];
        }
        acc = acc * sc2[c] + sh2[c];
        xf[(b * SSZ + s) * HSZ + c] = fmaxf(acc, 0.f);
    }
}

// ---------------------------------------------------------------------------
// Kernel 2: Wh = xf @ gat_W^T + gat_Wb via v_wmma_f32_16x16x32_f16,
//           s1 = Wh @ a1, s2 = Wh @ a2 via cross-lane shuffle reductions.
// ---------------------------------------------------------------------------
__global__ __launch_bounds__(128)
void k_gat_proj(const float* __restrict__ xf, const float* __restrict__ gW,
                const float* __restrict__ gWb, const float* __restrict__ a1,
                const float* __restrict__ a2,
                float* __restrict__ Wh, float* __restrict__ s1,
                float* __restrict__ s2) {
    const int lane  = threadIdx.x & 31;
    const int wave  = threadIdx.x >> 5;
    const int iBase = (blockIdx.x * 4 + wave) * 16;
    const int m     = lane & 15;
    const int g     = lane >> 4;

    const float* row = xf + (iBase + m) * HSZ;

    DFrag d[4];
    #pragma unroll
    for (int t = 0; t < 4; ++t)
        #pragma unroll
        for (int r = 0; r < 8; ++r) d[t].e[r] = 0.f;

    #pragma unroll
    for (int kb = 0; kb < HSZ; kb += 32) {
        AFrag a;                              // 16x32 A tile (ISA 7.12.2)
        #pragma unroll
        for (int h = 0; h < 16; ++h) {
            int k = kb + g * 16 + h;
            a.e[h] = (_Float16)row[k];
        }
        #pragma unroll
        for (int t = 0; t < 4; ++t) {         // B[k][n] = gat_W[cBase+n][k]
            AFrag bf;
            const float* wrow = gW + (t * 16 + m) * HSZ + kb + g * 16;
            #pragma unroll
            for (int h = 0; h < 16; ++h) bf.e[h] = (_Float16)wrow[h];
            d[t].v = __builtin_amdgcn_wmma_f32_16x16x32_f16(
                false, a.v, false, bf.v, (short)0, d[t].v, false, false);
        }
    }

    float s1acc[8], s2acc[8];
    #pragma unroll
    for (int r = 0; r < 8; ++r) { s1acc[r] = 0.f; s2acc[r] = 0.f; }

    #pragma unroll
    for (int t = 0; t < 4; ++t) {
        int c = t * 16 + m;
        float bias = gWb[c], a1c = a1[c], a2c = a2[c];
        #pragma unroll
        for (int r = 0; r < 8; ++r) {
            int M = r + 8 * g;
            float val = d[t].e[r] + bias;
            Wh[(iBase + M) * HSZ + c] = val;
            float p1 = val * a1c, p2 = val * a2c;
            #pragma unroll
            for (int off = 1; off < 16; off <<= 1) {
                p1 += __shfl_xor(p1, off, 32);
                p2 += __shfl_xor(p2, off, 32);
            }
            s1acc[r] += p1; s2acc[r] += p2;
        }
    }
    if (m == 0) {
        #pragma unroll
        for (int r = 0; r < 8; ++r) {
            s1[iBase + r + 8 * g] = s1acc[r];
            s2[iBase + r + 8 * g] = s2acc[r];
        }
    }
}

// ---------------------------------------------------------------------------
// Kernel 3: WhT[c][n] = (f16)Wh[n][c];  Kernel 3b: max(s2)
// ---------------------------------------------------------------------------
__global__ __launch_bounds__(256)
void k_transpose(const float* __restrict__ Wh, _Float16* __restrict__ WhT) {
    int idx = blockIdx.x * 256 + threadIdx.x;
    int n = idx >> 6, c = idx & 63;
    WhT[c * NSZ + n] = (_Float16)Wh[idx];
}

__global__ __launch_bounds__(256)
void k_s2max(const float* __restrict__ s2, float* __restrict__ out) {
    __shared__ float red[256];
    int tid = threadIdx.x;
    float mv = -3.4e38f;
    for (int i = tid; i < NSZ; i += 256) mv = fmaxf(mv, s2[i]);
    red[tid] = mv;
    __syncthreads();
    for (int s = 128; s > 0; s >>= 1) {
        if (tid < s) red[tid] = fmaxf(red[tid], red[tid + s]);
        __syncthreads();
    }
    if (tid == 0) out[0] = red[0];
}

// ---------------------------------------------------------------------------
// Kernel 4: fused GAT attention, split-K over j (4 chunks).
//   P_ij = exp(lrelu(s1_i + s2_j + ab) - rowmax_i)  [rowmax analytic]
//   One wave: one 16-row tile x all 4 H-column tiles for one j chunk.
//   A built once per k-step -> 4 WMMAs. Partials written unnormalized;
//   fixed-order combine happens in k_tail (deterministic, no float atomics).
// ---------------------------------------------------------------------------
__global__ __launch_bounds__(128)
void k_attn(const float* __restrict__ s1, const float* __restrict__ s2,
            const float* __restrict__ s2max, const float* __restrict__ gab,
            const _Float16* __restrict__ WhT,
            float* __restrict__ hpPart, float* __restrict__ rsPart) {
    __shared__ float ls2[CHLEN];
    const int lane    = threadIdx.x & 31;
    const int wave    = threadIdx.x >> 5;
    const int chunk   = blockIdx.x & (CHUNKS - 1);
    const int rowTile = (blockIdx.x >> 2) * 4 + wave;
    const int iBase   = rowTile * 16;
    const int m       = lane & 15;
    const int g       = lane >> 4;
    const int j0      = chunk * CHLEN;

    // cooperative stage of this chunk's s2 slice (8 KB)
    for (int i = threadIdx.x; i < CHLEN / 4; i += 128)
        ((float4*)ls2)[i] = ((const float4*)(s2 + j0))[i];
    __syncthreads();

    const float c0 = s1[iBase + m] + gab[0];
    float mx = c0 + s2max[0];
    mx = mx > 0.f ? mx : 0.2f * mx;            // analytic row max of e

    const _Float16* brow = WhT + m * NSZ + g * 16 + j0;

    DFrag d[4];
    #pragma unroll
    for (int t = 0; t < 4; ++t)
        #pragma unroll
        for (int r = 0; r < 8; ++r) d[t].e[r] = 0.f;
    float asum = 0.f;

    for (int jb = 0; jb < CHLEN; jb += 32) {
        AFrag a;
        const float* s2p = ls2 + jb + g * 16;
        #pragma unroll
        for (int h = 0; h < 8; ++h) {          // K = g*16 + h
            float t = c0 + s2p[h];
            t = t > 0.f ? t : 0.2f * t;
            float p = __expf(t - mx);
            asum += p;
            a.e[h] = (_Float16)p;
        }
        #pragma unroll
        for (int h = 0; h < 8; ++h) {          // K = g*16 + 8 + h
            float t = c0 + s2p[8 + h];
            t = t > 0.f ? t : 0.2f * t;
            float p = __expf(t - mx);
            asum += p;
            a.e[8 + h] = (_Float16)p;
        }
        __builtin_prefetch((const void*)(brow + jb + 128), 0, 0);
        #pragma unroll
        for (int t = 0; t < 4; ++t) {          // 4 H-column tiles per A build
            AFrag bf;
            bf.v = *(const v16h*)(brow + t * 16 * NSZ + jb);
            d[t].v = __builtin_amdgcn_wmma_f32_16x16x32_f16(
                false, a.v, false, bf.v, (short)0, d[t].v, false, false);
        }
    }

    float rsum = asum + __shfl_xor(asum, 16, 32);
    #pragma unroll
    for (int t = 0; t < 4; ++t)
        #pragma unroll
        for (int r = 0; r < 8; ++r) {
            int M = r + 8 * g;
            hpPart[((size_t)chunk * NSZ + iBase + M) * HSZ + t * 16 + m] =
                d[t].e[r];
        }
    if (g == 0) rsPart[chunk * NSZ + iBase + m] = rsum;
}

// ---------------------------------------------------------------------------
// Kernel 5: combine split-K partials (fixed order), normalize, then
// time-attention softmax, mean pool, 2-layer MLP head.
// ---------------------------------------------------------------------------
__global__ __launch_bounds__(128)
void k_tail(const float* __restrict__ hpPart, const float* __restrict__ rsPart,
            const float* __restrict__ taW, const float* __restrict__ tab,
            const float* __restrict__ p1W, const float* __restrict__ p1b,
            const float* __restrict__ p2W, const float* __restrict__ p2b,
            float* __restrict__ out) {
    __shared__ float pooled[HSZ];
    __shared__ float q1[32];
    const int tid = threadIdx.x;   // = s
    const int b   = blockIdx.x;
    const int n   = b * SSZ + tid;

    if (tid < HSZ) pooled[tid] = 0.f;
    __syncthreads();

    float rs = 0.f;
    #pragma unroll
    for (int ch = 0; ch < CHUNKS; ++ch) rs += rsPart[ch * NSZ + n];
    const float invrs = 1.f / rs;

    float hrow[HSZ];
    #pragma unroll 4
    for (int c = 0; c < HSZ; ++c) {
        float v = 0.f;
        #pragma unroll
        for (int ch = 0; ch < CHUNKS; ++ch)
            v += hpPart[((size_t)ch * NSZ + n) * HSZ + c];
        hrow[c] = v * invrs;
    }

    float tw[HSZ];
    float mx = -3.4e38f;
    for (int c = 0; c < HSZ; ++c) {
        float acc = tab[c];
        const float* wr = taW + c * HSZ;
        #pragma unroll 8
        for (int k = 0; k < HSZ; ++k) acc += hrow[k] * wr[k];
        tw[c] = acc;
        mx = fmaxf(mx, acc);
    }
    float se = 0.f;
    #pragma unroll 8
    for (int c = 0; c < HSZ; ++c) { tw[c] = __expf(tw[c] - mx); se += tw[c]; }
    float inv = (1.f / se) * (1.f / (float)SSZ);
    for (int c = 0; c < HSZ; ++c) atomicAdd(&pooled[c], hrow[c] * tw[c] * inv);
    __syncthreads();

    if (tid < 32) {
        float acc = p1b[tid];
        const float* wr = p1W + tid * HSZ;
        #pragma unroll 8
        for (int c = 0; c < HSZ; ++c) acc += pooled[c] * wr[c];
        q1[tid] = fmaxf(acc, 0.f);
    }
    __syncthreads();
    if (tid == 0) {
        float acc = p2b[0];
        #pragma unroll
        for (int j = 0; j < 32; ++j) acc += q1[j] * p2W[j];
        out[b] = acc;
    }
}

// ---------------------------------------------------------------------------
extern "C" void kernel_launch(void* const* d_in, const int* in_sizes, int n_in,
                              void* d_out, int out_size, void* d_ws,
                              size_t ws_size, hipStream_t stream) {
    const float* x    = (const float*)d_in[0];
    const float* c1w  = (const float*)d_in[1];
    const float* c1b  = (const float*)d_in[2];
    const float* g1   = (const float*)d_in[3];
    const float* b1   = (const float*)d_in[4];
    const float* m1   = (const float*)d_in[5];
    const float* v1   = (const float*)d_in[6];
    const float* c2w  = (const float*)d_in[7];
    const float* c2b  = (const float*)d_in[8];
    const float* g2   = (const float*)d_in[9];
    const float* b2   = (const float*)d_in[10];
    const float* m2   = (const float*)d_in[11];
    const float* v2   = (const float*)d_in[12];
    const float* gW   = (const float*)d_in[13];
    const float* gWb  = (const float*)d_in[14];
    const float* ga1  = (const float*)d_in[15];
    const float* ga2  = (const float*)d_in[16];
    const float* gab  = (const float*)d_in[17];
    const float* taW  = (const float*)d_in[18];
    const float* tab  = (const float*)d_in[19];
    const float* p1W  = (const float*)d_in[20];
    const float* p1b  = (const float*)d_in[21];
    const float* p2W  = (const float*)d_in[22];
    const float* p2b  = (const float*)d_in[23];
    float* out = (float*)d_out;

    char* w = (char*)d_ws;
    float*     xf     = (float*)(w);                       // 2 MiB
    float*     Wh     = (float*)(w + (size_t)2097152);     // 2 MiB
    _Float16*  WhT    = (_Float16*)(w + (size_t)4194304);  // 1 MiB (32B aligned)
    float*     s1     = (float*)(w + (size_t)5242880);     // 32 KiB
    float*     s2     = (float*)(w + (size_t)5275648);     // 32 KiB
    float*     s2mx   = (float*)(w + (size_t)5308416);     // 256 B
    float*     rsPart = (float*)(w + (size_t)5308672);     // 128 KiB
    float*     hpPart = (float*)(w + (size_t)5439744);     // 8 MiB

    k_feat<<<BSZ, 256, 0, stream>>>(x, c1w, c1b, g1, b1, m1, v1,
                                    c2w, c2b, g2, b2, m2, v2, xf);
    k_gat_proj<<<NSZ / 64, 128, 0, stream>>>(xf, gW, gWb, ga1, ga2, Wh, s1, s2);
    k_transpose<<<(NSZ * HSZ) / 256, 256, 0, stream>>>(Wh, WhT);
    k_s2max<<<1, 256, 0, stream>>>(s2, s2mx);
    k_attn<<<(NSZ / 16 / 4) * CHUNKS, 128, 0, stream>>>(s1, s2, s2mx, gab,
                                                        WhT, hpPart, rsPart);
    k_tail<<<BSZ, 128, 0, stream>>>(hpPart, rsPart, taW, tab,
                                    p1W, p1b, p2W, p2b, out);
}
